// PlyBitLinear_69630009802893
// MI455X (gfx1250) — compile-verified, matrix operand order checked by reading, not run on hardware
//
#include <hip/hip_runtime.h>

// BitLinear ternary GEMM for gfx1250 (MI455X):
//   C[M,N] (f32) = (f16)(A[M,K] f16 x dequant(Wpacked)[K,N]) * scale
// Compute-bound: maximize V_WMMA_F32_16X16X32_F16 issue; dequant via LDS LUT.

typedef __attribute__((ext_vector_type(16))) _Float16 v16h;
typedef __attribute__((ext_vector_type(8)))  _Float16 v8h;
typedef __attribute__((ext_vector_type(8)))  float    v8f;

#define MDIM 8192
#define KDIM 4096
#define NDIM 4096

#define BM 128
#define BN 256
#define BK 32
#define NKT (KDIM / BK)   // 128 K-steps
#define LDA 40            // BK + 8 halves pad: 16B-aligned rows, bank spread
#define LDB 40

union V16Pair { struct { v8h lo, hi; } p; v16h v; };

__global__ __launch_bounds__(256)
void bitlinear_wmma_f16(const _Float16* __restrict__ A,
                        const unsigned char* __restrict__ W,
                        const float* __restrict__ scale_p,
                        float* __restrict__ C)
{
    // double-buffered tiles + dequant LUT: 20480 + 40960 + 2048 = 63488 B
    __shared__ _Float16 As[2][BM][LDA];           // [m][k]
    __shared__ _Float16 Bs[2][BN][LDB];           // [n][k] (contiguous K per column)
    __shared__ unsigned long long lut[256];       // byte -> 4 packed fp16 {0,+1,-1}

    const int tid  = threadIdx.x;
    const int lane = tid & 31;
    const int wave = tid >> 5;
    const int wm   = wave & 1;    // 2 slabs of 64 rows
    const int wn   = wave >> 1;   // 4 slabs of 64 cols

    const int m0 = blockIdx.y * BM;
    const int n0 = blockIdx.x * BN;
    const float scale = *scale_p;

    // ---- build ternary dequant LUT (one entry per thread, once) ----
    {
        unsigned int b = tid;                     // blockDim.x == 256
        unsigned long long hh = 0ull;
        #pragma unroll
        for (int q = 0; q < 4; ++q) {
            unsigned int code = (b >> (2 * q)) & 3u;
            unsigned long long h =
                (code == 1u) ? 0x3C00ull : (code == 2u) ? 0xBC00ull : 0ull;
            hh |= h << (16 * q);
        }
        lut[b] = hh;
    }

    // fp32 accumulators: 4 (M) x 4 (N) tiles of 16x16 per wave
    v8f acc[4][4];
    #pragma unroll
    for (int i = 0; i < 4; ++i)
        #pragma unroll
        for (int j = 0; j < 4; ++j)
            #pragma unroll
            for (int r = 0; r < 8; ++r)
                acc[i][j][r] = 0.0f;

    const int halfsel = (lane >> 4) * 8;  // lanes 0-15: K 0..7/16..23; 16-31: 8..15/24..31
    const int lr      = lane & 15;

    // staging registers (software pipeline of global loads)
    v8h aReg[2];
    unsigned int bReg[2];

    const int a_row = tid >> 2;           // 0..63 (x2 via i)
    const int a_col = (tid & 3) * 8;      // 0,8,16,24
    const int b_kpl = tid >> 5;           // packed-K row 0..7
    const int b_nl  = (tid & 31) * 8;     // column group 0..248

    auto load_tile = [&](int kt) {
        const int kb = kt * BK;
        #pragma unroll
        for (int i = 0; i < 2; ++i) {
            int row = a_row + i * 64;
            aReg[i] = *(const v8h*)(A + (size_t)(m0 + row) * KDIM + kb + a_col);
        }
        const unsigned char* gp =
            W + (size_t)(kb / 4 + b_kpl) * NDIM + n0 + b_nl;   // 8B aligned
        bReg[0] = ((const unsigned int*)gp)[0];
        bReg[1] = ((const unsigned int*)gp)[1];
    };

    auto stage_tile = [&](int buf) {
        #pragma unroll
        for (int i = 0; i < 2; ++i)
            *(v8h*)&As[buf][a_row + i * 64][a_col] = aReg[i];
        #pragma unroll
        for (int h = 0; h < 2; ++h) {
            unsigned int pw = bReg[h];
            #pragma unroll
            for (int j = 0; j < 4; ++j) {
                unsigned int byte = (pw >> (8 * j)) & 0xFFu;
                // one ds_load_b64 (LUT) + one ds_store_b64 per byte
                *(unsigned long long*)&Bs[buf][b_nl + 4 * h + j][b_kpl * 4] =
                    lut[byte];
            }
        }
    };

    load_tile(0);   // prologue

    for (int kt = 0; kt < NKT; ++kt) {
        const int buf = kt & 1;

        stage_tile(buf);
        __syncthreads();                       // one barrier per K-step
        if (kt + 1 < NKT) load_tile(kt + 1);   // overlap globals with WMMAs

        // keep L2 warm for the tile after next (global_prefetch_b8)
        if (kt + 2 < NKT) {
            __builtin_prefetch(A + (size_t)(m0 + a_row) * KDIM + (kt + 2) * BK, 0, 1);
        }

        // hoist ALL 8 fragments into distinct registers so the 16 frag
        // ds_load_b128s issue up-front and dscnt waits stagger across WMMAs
        v16h bfrag[4];
        #pragma unroll
        for (int j = 0; j < 4; ++j) {
            int col = wn * 64 + j * 16 + lr;
            V16Pair t;
            t.p.lo = *(const v8h*)&Bs[buf][col][halfsel];
            t.p.hi = *(const v8h*)&Bs[buf][col][halfsel + 16];
            bfrag[j] = t.v;
        }
        v16h afrag[4];
        #pragma unroll
        for (int i = 0; i < 4; ++i) {
            int row = wm * 64 + i * 16 + lr;
            V16Pair t;
            t.p.lo = *(const v8h*)&As[buf][row][halfsel];
            t.p.hi = *(const v8h*)&As[buf][row][halfsel + 16];
            afrag[i] = t.v;
        }

        #pragma unroll
        for (int i = 0; i < 4; ++i)
            #pragma unroll
            for (int j = 0; j < 4; ++j)
                acc[i][j] = __builtin_amdgcn_wmma_f32_16x16x32_f16(
                    /*neg_a=*/false, afrag[i],
                    /*neg_b=*/false, bfrag[j],
                    /*c_mod=*/(short)0, acc[i][j],
                    /*reuse_a=*/false, /*reuse_b=*/false);
    }

    // ---- epilogue: cast to f16 (reference rounding), scale in f32, store ----
    #pragma unroll
    for (int i = 0; i < 4; ++i) {
        #pragma unroll
        for (int j = 0; j < 4; ++j) {
            int gn    = n0 + wn * 64 + j * 16 + lr;
            int mbase = m0 + wm * 64 + i * 16 + halfsel;  // +8 rows for lanes 16-31
            #pragma unroll
            for (int r = 0; r < 8; ++r) {
                C[(size_t)(mbase + r) * NDIM + gn] =
                    (float)(_Float16)acc[i][j][r] * scale;
            }
        }
    }
}

extern "C" void kernel_launch(void* const* d_in, const int* in_sizes, int n_in,
                              void* d_out, int out_size, void* d_ws, size_t ws_size,
                              hipStream_t stream) {
    (void)in_sizes; (void)n_in; (void)out_size; (void)d_ws; (void)ws_size;
    const _Float16*      x     = (const _Float16*)d_in[0];
    const unsigned char* wpk   = (const unsigned char*)d_in[1];
    const float*         scale = (const float*)d_in[2];
    float*               out   = (float*)d_out;

    dim3 grid(NDIM / BN, MDIM / BM);   // 16 x 64 workgroups
    dim3 block(256);                   // 8 wave32s
    bitlinear_wmma_f16<<<grid, block, 0, stream>>>(x, wpk, scale, out);
}
